// LPUR_20340965114022
// MI455X (gfx1250) — compile-verified, matrix operand order checked by reading, not run on hardware
//
#include <hip/hip_runtime.h>

// Problem constants (from reference setup_inputs)
#define BATCH 8
#define C_    256
#define HW_   16384        // 128*128
#define NPIX_ 131072       // BATCH*HW_
#define R_    256
#define O_    256

#define LSTR  264          // padded LDS row stride (elements): conflict-free 16B frag reads
#define WAVES 4
#define PIX_PER_BLOCK 256  // 4 waves * 4 tiles * 16 rows

typedef __attribute__((ext_vector_type(16))) __bf16         v16bf;
typedef __attribute__((ext_vector_type(8)))  float          f32x8;
typedef __attribute__((ext_vector_type(8)))  unsigned short u16x8;

// Hardware bf16 convert (native v_cvt on gfx1250) instead of manual integer RNE.
__device__ __forceinline__ unsigned short f2bf(float f) {
  __bf16 h = (__bf16)f;
  return __builtin_bit_cast(unsigned short, h);
}
__device__ __forceinline__ float bf2f(unsigned short h) {
  return __uint_as_float(((unsigned int)h) << 16);
}

// Load a 16x32 bf16 fragment (A layout; also used for B as rows of B^T).
// lane<16 : row = row0+lane,    K = {k0..k0+7, k0+16..k0+23}
// lane>=16: row = row0+lane-16, K = {k0+8..k0+15, k0+24..k0+31}
__device__ __forceinline__ v16bf load_frag(const unsigned short* base, int stride,
                                           int row0, int k0, int lane) {
  const int r  = row0 + (lane & 15);
  const int ko = (lane < 16) ? 0 : 8;
  const unsigned short* p = base + r * stride + k0 + ko;
  union { v16bf v; u16x8 h[2]; } f;
  f.h[0] = *(const u16x8*)(p);
  f.h[1] = *(const u16x8*)(p + 16);
  return f.v;
}

__device__ __forceinline__ f32x8 wmma_bf16(v16bf a, v16bf b, f32x8 c) {
  return __builtin_amdgcn_wmma_f32_16x16x32_bf16(false, a, false, b, (short)0, c, false, false);
}

// K=256 GEMM micro-kernel with double-buffered B fragments so WMMA overlaps
// the next fragment's ds/global loads (avoids full s_wait_dscnt drains).
__device__ __forceinline__ f32x8 gemm8(const v16bf a[8], const unsigned short* base,
                                       int stride, int row0, int lane, f32x8 acc) {
  v16bf b0 = load_frag(base, stride, row0, 0, lane);
  v16bf b1 = load_frag(base, stride, row0, 32, lane);
  #pragma unroll
  for (int ks = 0; ks < 8; ks += 2) {
    acc = wmma_bf16(a[ks], b0, acc);
    if (ks + 2 < 8) b0 = load_frag(base, stride, row0, (ks + 2) * 32, lane);
    acc = wmma_bf16(a[ks + 1], b1, acc);
    if (ks + 3 < 8) b1 = load_frag(base, stride, row0, (ks + 3) * 32, lane);
  }
  return acc;
}

// ---------------------------------------------------------------------------
// Kernel 1: convert token bank + weights to bf16 layouts, zero BN accumulators
// ---------------------------------------------------------------------------
__global__ void prep_kernel(const float* __restrict__ t_items,
                            const float* __restrict__ w_out,
                            unsigned short* __restrict__ tb,
                            unsigned short* __restrict__ tTb,
                            unsigned short* __restrict__ w1b,
                            unsigned short* __restrict__ w2b,
                            float* __restrict__ sums,
                            float* __restrict__ sumsq) {
  const int r = blockIdx.x;
  const int c = threadIdx.x;
  const unsigned short tv = f2bf(t_items[r * C_ + c]);
  tb [r * C_ + c] = tv;        // [R, C] row-major (B^T rows for GEMM-1)
  tTb[c * R_ + r] = tv;        // [C, R] row-major (B^T rows for GEMM-2)
  w1b[r * C_ + c] = f2bf(w_out[r * 2 * C_ + c]);        // W[:, :C]
  w2b[r * C_ + c] = f2bf(w_out[r * 2 * C_ + C_ + c]);   // W[:, C:]
  if (blockIdx.x == 0) { sums[c] = 0.f; sumsq[c] = 0.f; }
}

// ---------------------------------------------------------------------------
// Kernel 2: per-pixel L2 normalize over channels, emit bf16 qn[N, C]
// ---------------------------------------------------------------------------
__global__ void normalize_kernel(const float* __restrict__ q,
                                 unsigned short* __restrict__ qn) {
  const int p  = blockIdx.x * 256 + threadIdx.x;
  const int b  = p >> 14;
  const int hw = p & (HW_ - 1);
  const float* src = q + b * C_ * HW_ + hw;   // channel stride HW_, lane-coalesced
  float acc = 0.f;
  #pragma unroll 8
  for (int c = 0; c < C_; ++c) { float v = src[c * HW_]; acc += v * v; }
  const float rn = 1.f / fmaxf(sqrtf(acc), 1e-12f);
  unsigned int* dst = (unsigned int*)qn + p * (C_ / 2);
  #pragma unroll 8
  for (int c = 0; c < C_; c += 2) {
    float v0 = src[c * HW_] * rn;
    float v1 = src[(c + 1) * HW_] * rn;
    dst[c >> 1] = (unsigned int)f2bf(v0) | ((unsigned int)f2bf(v1) << 16);
  }
}

// ---------------------------------------------------------------------------
// Kernel 3: fused cue GEMM -> softmax -> read GEMM -> output GEMM (+BN stats)
// block = 128 threads (4 waves), 256 pixels/block.
// cue/read tiles stream through per-wave LDS scratch (no 128-VGPR arrays).
// ---------------------------------------------------------------------------
__global__ void lpur_main_kernel(const unsigned short* __restrict__ qn,
                                 const unsigned short* __restrict__ tb,
                                 const unsigned short* __restrict__ tTb,
                                 const unsigned short* __restrict__ w1b,
                                 const unsigned short* __restrict__ w2b,
                                 float* __restrict__ y,
                                 float* __restrict__ sums,
                                 float* __restrict__ sumsq) {
  extern __shared__ char smem[];
  unsigned short* tbS  = (unsigned short*)smem;                         // 256*264
  unsigned short* tTbS = (unsigned short*)(smem + 256 * LSTR * 2);      // 256*264
  unsigned short* scr  = (unsigned short*)(smem + 2 * 256 * LSTR * 2);  // WAVES*16*264

  const int tid  = threadIdx.x;
  const int lane = tid & 31;
  const int wv   = tid >> 5;
  const int col  = lane & 15;
  const int rh   = (lane < 16) ? 0 : 8;

  // ---- stage token bank (and transpose) into LDS, 16B chunks, coalesced ----
  for (int idx = tid; idx < 256 * 32; idx += 128) {
    const int row = idx >> 5;
    const int ch  = (idx & 31) * 8;
    *(u16x8*)(tbS  + row * LSTR + ch) = *(const u16x8*)(tb  + row * C_ + ch);
    *(u16x8*)(tTbS + row * LSTR + ch) = *(const u16x8*)(tTb + row * R_ + ch);
  }
  __syncthreads();

  unsigned short* sc = scr + wv * (16 * LSTR);

  for (int t = 0; t < 4; ++t) {
    const int p0 = blockIdx.x * PIX_PER_BLOCK + wv * 64 + t * 16;

    // A fragments of the normalized query (kept live through GEMM-3)
    v16bf qA[8];
    #pragma unroll
    for (int ks = 0; ks < 8; ++ks)
      qA[ks] = load_frag(qn + p0 * C_, C_, 0, ks * 32, lane);

    // ---- GEMM-1: cue tile -> bf16 scratch; track running row-max in regs ----
    float mx[8];
    #pragma unroll
    for (int v = 0; v < 8; ++v) mx[v] = -3.0e38f;
    #pragma unroll
    for (int nt = 0; nt < 16; ++nt) {
      f32x8 acc = {};
      acc = gemm8(qA, tbS, LSTR, nt * 16, lane, acc);
      #pragma unroll
      for (int v = 0; v < 8; ++v) {
        mx[v] = fmaxf(mx[v], acc[v]);
        sc[(v + rh) * LSTR + nt * 16 + col] = f2bf(acc[v]);
      }
    }
    #pragma unroll
    for (int v = 0; v < 8; ++v) {
      float m = mx[v];
      m = fmaxf(m, __shfl_xor(m, 1, 32));
      m = fmaxf(m, __shfl_xor(m, 2, 32));
      m = fmaxf(m, __shfl_xor(m, 4, 32));
      m = fmaxf(m, __shfl_xor(m, 8, 32));
      mx[v] = m;
    }

    // ---- softmax pass 2: in-place exp on scratch + row sums ----
    float sm[8];
    #pragma unroll
    for (int v = 0; v < 8; ++v) sm[v] = 0.f;
    #pragma unroll
    for (int nt = 0; nt < 16; ++nt)
      #pragma unroll
      for (int v = 0; v < 8; ++v) {
        const int idx = (v + rh) * LSTR + nt * 16 + col;
        const float e = __expf(bf2f(sc[idx]) - mx[v]);
        sm[v] += e;
        sc[idx] = f2bf(e);   // unscaled exp in [0,1]; 1/sum folded into GEMM-2
      }
    #pragma unroll
    for (int v = 0; v < 8; ++v) {
      float s = sm[v];
      s += __shfl_xor(s, 1, 32);
      s += __shfl_xor(s, 2, 32);
      s += __shfl_xor(s, 4, 32);
      s += __shfl_xor(s, 8, 32);
      sm[v] = 1.f / s;
    }
    __syncthreads();

    // attn in A-layout
    v16bf aA[8];
    #pragma unroll
    for (int ks = 0; ks < 8; ++ks) aA[ks] = load_frag(sc, LSTR, 0, ks * 32, lane);
    __syncthreads();

    // ---- GEMM-2: read tile = attn x T; scale by 1/sum; straight to scratch ----
    #pragma unroll
    for (int ct = 0; ct < 16; ++ct) {
      f32x8 acc = {};
      acc = gemm8(aA, tTbS, LSTR, ct * 16, lane, acc);
      #pragma unroll
      for (int v = 0; v < 8; ++v)
        sc[(v + rh) * LSTR + ct * 16 + col] = f2bf(acc[v] * sm[v]);
    }
    __syncthreads();

    v16bf rA[8];
    #pragma unroll
    for (int ks = 0; ks < 8; ++ks) rA[ks] = load_frag(sc, LSTR, 0, ks * 32, lane);
    __syncthreads();

    // ---- GEMM-3: y = qn*W1^T + read*W2^T ; emit y + BN partial stats ----
    #pragma unroll
    for (int ot = 0; ot < 16; ++ot) {
      f32x8 acc = {};
      acc = gemm8(qA, w1b, C_, ot * 16, lane, acc);
      acc = gemm8(rA, w2b, C_, ot * 16, lane, acc);

      float s1 = 0.f, s2 = 0.f;
      #pragma unroll
      for (int v = 0; v < 8; ++v) {
        const float val = acc[v];
        y[(p0 + v + rh) * O_ + ot * 16 + col] = val;
        s1 += val;
        s2 += val * val;
      }
      s1 += __shfl_xor(s1, 16, 32);
      s2 += __shfl_xor(s2, 16, 32);
      if (lane < 16) {
        atomicAdd(&sums [ot * 16 + col], s1);
        atomicAdd(&sumsq[ot * 16 + col], s2);
      }
    }
  }
}

// ---------------------------------------------------------------------------
// Kernel 4: BatchNorm(batch stats) + ReLU, with LDS transpose [N,O]->[B,C,H,W]
// ---------------------------------------------------------------------------
#define TSTR 261   // padded f32 tile stride (odd mod-64 -> conflict-free column reads)
__global__ void bn_finalize_kernel(const float* __restrict__ y,
                                   const float* __restrict__ sums,
                                   const float* __restrict__ sumsq,
                                   const float* __restrict__ gamma,
                                   const float* __restrict__ beta,
                                   float* __restrict__ out) {
  extern __shared__ char fsm[];
  float* tileS  = (float*)fsm;           // 64 x TSTR
  float* scaleS = tileS + 64 * TSTR;     // 256
  float* shiftS = scaleS + 256;          // 256

  const int t = threadIdx.x;
  const int pbase = blockIdx.x * 64;

  {
    const float inv_n = 1.f / (float)NPIX_;
    const float mean  = sums[t] * inv_n;
    const float var   = fmaxf(sumsq[t] * inv_n - mean * mean, 0.f);
    const float sc    = gamma[t] * rsqrtf(var + 1e-5f);
    scaleS[t] = sc;
    shiftS[t] = beta[t] - mean * sc;
  }
  for (int i = 0; i < 64; ++i)
    tileS[i * TSTR + t] = y[(pbase + i) * O_ + t];    // coalesced 1KB rows
  __syncthreads();

  const int b   = pbase >> 14;
  const int hw0 = pbase & (HW_ - 1);
  for (int j = 0; j < 64; ++j) {
    const int idx = j * 256 + t;
    const int o   = idx >> 6;
    const int px  = idx & 63;
    const float v = tileS[px * TSTR + o] * scaleS[o] + shiftS[o];
    out[(b * O_ + o) * HW_ + hw0 + px] = fmaxf(v, 0.f);
  }
}

// ---------------------------------------------------------------------------
extern "C" void kernel_launch(void* const* d_in, const int* in_sizes, int n_in,
                              void* d_out, int out_size, void* d_ws, size_t ws_size,
                              hipStream_t stream) {
  const float* query   = (const float*)d_in[0];
  const float* t_items = (const float*)d_in[1];
  const float* w_out   = (const float*)d_in[2];
  const float* gamma   = (const float*)d_in[3];
  const float* beta    = (const float*)d_in[4];
  float* out = (float*)d_out;

  // workspace carve-up
  char* ws = (char*)d_ws;
  size_t off = 0;
  auto carve = [&](size_t bytes) { void* p = ws + off; off = (off + bytes + 255) & ~(size_t)255; return p; };
  unsigned short* qn    = (unsigned short*)carve((size_t)NPIX_ * C_ * 2);  // 64 MB
  unsigned short* tb    = (unsigned short*)carve((size_t)R_ * C_ * 2);
  unsigned short* tTb   = (unsigned short*)carve((size_t)C_ * R_ * 2);
  unsigned short* w1b   = (unsigned short*)carve((size_t)O_ * C_ * 2);
  unsigned short* w2b   = (unsigned short*)carve((size_t)O_ * C_ * 2);
  float*          ybuf  = (float*)carve((size_t)NPIX_ * O_ * 4);           // 128 MB
  float*          sums  = (float*)carve(O_ * 4);
  float*          sumsq = (float*)carve(O_ * 4);

  prep_kernel<<<R_, C_, 0, stream>>>(t_items, w_out, tb, tTb, w1b, w2b, sums, sumsq);
  normalize_kernel<<<NPIX_ / 256, 256, 0, stream>>>(query, qn);

  const size_t main_lds = (size_t)(2 * 256 * LSTR * 2 + WAVES * 16 * LSTR * 2); // 304128 B
  lpur_main_kernel<<<NPIX_ / PIX_PER_BLOCK, WAVES * 32, main_lds, stream>>>(
      qn, tb, tTb, w1b, w2b, ybuf, sums, sumsq);

  const size_t fin_lds = (size_t)(64 * TSTR + 512) * 4;                         // 68864 B
  bn_finalize_kernel<<<NPIX_ / 64, 256, fin_lds, stream>>>(ybuf, sums, sumsq, gamma, beta, out);
}
